// PolaLinearAttention_4355096838554
// MI455X (gfx1250) — matrix operand order repaired and checked by reading, not compiled, not run on hardware
//
#include <hip/hip_runtime.h>
#include <hip/hip_bf16.h>
#include <stdint.h>

// ---------------------------------------------------------------------------
// PolaLinearAttention for MI455X (gfx1250).  bf16 WMMA GEMMs with fused
// epilogues; linear-attention core collapsed to one batched WMMA GEMM per
// (b,h) against a tiny precomputed 80x128 matrix.  LDS tiles are streamed
// with GLOBAL_LOAD_ASYNC_TO_LDS_B128 (ASYNCcnt) and double-buffered.
// ---------------------------------------------------------------------------

typedef __bf16 bf16;
typedef __attribute__((ext_vector_type(16))) __bf16 v16bf;
typedef __attribute__((ext_vector_type(8)))  __bf16 v8bf;
typedef __attribute__((ext_vector_type(8)))  float  v8f;
typedef __attribute__((ext_vector_type(4)))  unsigned int u32x4;

#define NUM_HEADS 12
#define HD        64
#define CDIM      768
#define NSEQ      4096
#define BDIM      8
#define MROWS     (BDIM * NSEQ)      // 32768
#define EPSV      1e-6f

// ---------------- small device helpers ----------------
__device__ __forceinline__ bf16 f2bf(float f) {
  union { float f; uint32_t u; } c; c.f = f;
  uint32_t u = c.u;
  uint32_t r = (u + 0x7FFFu + ((u >> 16) & 1u)) >> 16;   // round-nearest-even
  union { unsigned short s; bf16 b; } o; o.s = (unsigned short)r;
  return o.b;
}
__device__ __forceinline__ float bf2f(bf16 b) {
  union { unsigned short s; bf16 b; } i; i.b = b;
  union { uint32_t u; float f; } o; o.u = ((uint32_t)i.s) << 16;
  return o.f;
}
__device__ __forceinline__ float softplus_f(float x) {
  return (x > 20.f) ? x : log1pf(__expf(x));
}
__device__ __forceinline__ float sigmoid_f(float x) {
  return 1.f / (1.f + __expf(-x));
}
__device__ __forceinline__ float safe_pow_f(float x, float p) {
  return (x > 0.f) ? __expf(p * __logf(x)) : 0.f;
}

// ---- CDNA5 async global->LDS copy (GVS mode: s[base] + v[off32]) ----------
// Generic pointers to LDS carry the LDS byte offset in their low 32 bits
// (flat aperture mapping).
__device__ __forceinline__ void async_copy_b128(const void* gbase, uint32_t gbyte_off,
                                                void* lds_ptr) {
  uint32_t lds_addr = (uint32_t)(uintptr_t)lds_ptr;
  asm volatile("global_load_async_to_lds_b128 %0, %1, %2"
               :: "v"(lds_addr), "v"(gbyte_off),
                  "s"((uint64_t)(uintptr_t)gbase)
               : "memory");
}
template <int N>
__device__ __forceinline__ void wait_asynccnt() {
  asm volatile("s_wait_asynccnt %0" :: "i"(N) : "memory");
}

// A-fragment (16x32 MxK bf16): lane<16 -> row M=lane, K {0..7,16..23};
// lane>=16 -> row M=lane-16, K {8..15,24..31}.  rowp points at K=0 of the row.
__device__ __forceinline__ v16bf load_afrag(const bf16* rowp, int klo) {
  union { v16bf v; v8bf h[2]; } u;
  u.h[0] = *(const v8bf*)(rowp + klo);
  u.h[1] = *(const v8bf*)(rowp + klo + 16);
  return u.v;
}
// B-fragment from Bt (N-major, K contiguous): lane<16 -> col N=lane, K {0..7,16..23};
// lane>=16 -> col N=lane-16, K {8..15,24..31}.
__device__ __forceinline__ v16bf load_bfrag(const bf16* rowp, int kb) {
  union { v16bf v; v8bf h[2]; } u;
  u.h[0] = *(const v8bf*)(rowp + kb / 2);
  u.h[1] = *(const v8bf*)(rowp + kb / 2 + 16);
  return u.v;
}

// ---------------- conversion kernels ----------------
__global__ __launch_bounds__(256) void cvt_f32_bf16_k(const float* __restrict__ in,
                                                      bf16* __restrict__ out, size_t n) {
  size_t i = (size_t)blockIdx.x * 256 + threadIdx.x;
  if (i < n) out[i] = f2bf(in[i]);
}
// in: (K x Ncols) row-major fp32  ->  out: (Ncols x K) row-major bf16
__global__ __launch_bounds__(256) void transpose_f32_bf16_k(const float* __restrict__ in,
                                                            bf16* __restrict__ out,
                                                            int K, int Ncols) {
  size_t i = (size_t)blockIdx.x * 256 + threadIdx.x;
  if (i >= (size_t)K * Ncols) return;
  int k = (int)(i / Ncols);
  int n = (int)(i % Ncols);
  out[(size_t)n * K + k] = f2bf(in[i]);
}

// ---------------- main WMMA GEMM: C = A(MxK) * Bt^T, fused epilogues -------
// block tile 128(M) x 64(N), 8 waves each own 16 rows x 64 cols, K-step 32.
// Double-buffered async LDS staging: next tile streams while WMMAs run.
struct GemmParams {
  const bf16* A;        // M x K row-major
  const bf16* Bt;       // Ncols x K row-major (transposed weights)
  int M, K, Ncols;
  bf16* outPos;         // qp / kp
  bf16* outNeg;         // qn / kn
  float* outF;          // g  / v
  float* outFinal;      // MODE 2 output
  const float* scale_p;
  const float* power_p;
  const float* pos_enc;
  const float* bias;
};

template <int MODE>   // 0 = x@Wqg -> qp,qn,g ; 1 = x@Wkv -> kp,kn,v ; 2 = proj
__global__ __launch_bounds__(256) void gemm_bf16_wmma(GemmParams p) {
  __shared__ __attribute__((aligned(16))) bf16 As[2][128][40];
  __shared__ __attribute__((aligned(16))) bf16 Bs[2][64][40];

  const int t    = threadIdx.x;
  const int lane = t & 31;
  const int wave = t >> 5;
  const int lc   = lane & 15;
  const int klo  = (lane & 16) ? 8 : 0;
  const int kb   = (lane & 16) ? 16 : 0;
  const size_t mBase = (size_t)blockIdx.x * 128;
  const int    nBase = blockIdx.y * 64;
  const int    nk    = p.K / 32;

  // issue async loads for K-tile ki into buffer buf (3 x b128 per thread)
  auto issue_tile = [&](int ki, int buf) {
    const int k0 = ki * 32;
    #pragma unroll
    for (int c0 = 0; c0 < 2; ++c0) {
      int ch = t + c0 * 256;                      // 512 A chunks of 16B
      int row = ch >> 2, q = ch & 3;
      async_copy_b128(p.A,
                      (uint32_t)(((mBase + row) * (size_t)p.K + k0 + q * 8) * 2),
                      &As[buf][row][q * 8]);
    }
    {
      int row = t >> 2, q = t & 3;                // 256 B chunks of 16B
      async_copy_b128(p.Bt,
                      (uint32_t)(((size_t)(nBase + row) * p.K + k0 + q * 8) * 2),
                      &Bs[buf][row][q * 8]);
    }
  };

  v8f acc[4] = {};
  issue_tile(0, 0);
  for (int ki = 0; ki < nk; ++ki) {
    const int buf = ki & 1;
    if (ki + 1 < nk) {
      issue_tile(ki + 1, buf ^ 1);   // overlap next tile with this one's WMMAs
      wait_asynccnt<3>();            // current tile done; next 3 still in flight
    } else {
      wait_asynccnt<0>();
    }
    __syncthreads();

    v16bf a = load_afrag(&As[buf][wave * 16 + lc][0], klo);
    #pragma unroll
    for (int j = 0; j < 4; ++j) {
      v16bf bb = load_bfrag(&Bs[buf][j * 16 + lc][0], kb);
      acc[j] = __builtin_amdgcn_wmma_f32_16x16x32_bf16(false, a, false, bb,
                                                       (short)0, acc[j], false, false);
    }
    __syncthreads();   // all reads done before this buffer is refilled
  }

  // fused epilogue
  #pragma unroll
  for (int j = 0; j < 4; ++j) {
    int col = nBase + j * 16 + lc;
    #pragma unroll
    for (int i = 0; i < 8; ++i) {
      size_t row = mBase + wave * 16 + ((lane >> 4) * 8) + i;
      float vacc = acc[j][i];
      if (MODE == 2) {
        p.outFinal[row * CDIM + col] = vacc + p.bias[col];
      } else if (col < CDIM) {
        float sc = softplus_f(p.scale_p[col]);
        float pw = 1.f + 4.f * sigmoid_f(p.power_p[col]);
        float val;
        if (MODE == 1) {
          int n = (int)(row & (NSEQ - 1));
          val = (vacc + p.pos_enc[(size_t)n * CDIM + col]) / sc;
        } else {
          val = vacc / sc;
        }
        size_t o = row * CDIM + col;
        p.outPos[o] = f2bf(safe_pow_f(val, pw));
        p.outNeg[o] = f2bf(safe_pow_f(-val, pw));
      } else {
        p.outF[row * CDIM + (col - CDIM)] = vacc;   // g or v (fp32)
      }
    }
  }
}

// ---------------- per-(b,h) kv / k_mean reduction -> Mt (80 x 128 bf16) ----
// Mt columns: [0..31]=kv1, [32..63]=rowswap(kv2), 64=k_mean, 65=rowswap(k_mean),
// [66..79]=0 (padding so the attention GEMM is 5 clean 16-col WMMA tiles).
__global__ __launch_bounds__(256) void kv_reduce_k(const bf16* __restrict__ kp,
                                                   const bf16* __restrict__ kn,
                                                   const float* __restrict__ v,
                                                   bf16* __restrict__ Mt) {
  __shared__ __attribute__((aligned(16))) bf16  kc[32][128];   // n x (kp||kn)
  __shared__ __attribute__((aligned(16))) float vv[32][64];    // n x v
  __shared__ __attribute__((aligned(16))) float kvs[128][64];
  __shared__ float ksum[128];

  const int bh = blockIdx.x;
  const int b = bh / NUM_HEADS, h = bh % NUM_HEADS;
  const int t = threadIdx.x;
  const int kk = t >> 1;            // 0..127
  const int vb = (t & 1) * 32;      // column half of v

  float acc[32];
  #pragma unroll
  for (int j = 0; j < 32; ++j) acc[j] = 0.f;
  float ks = 0.f;

  for (int n0 = 0; n0 < NSEQ; n0 += 32) {
    // kc cols: [0..63] from kp, [64..127] from kn -- bases kept wave-uniform
    {
      int r = t >> 3, q = t & 7;    // 256 chunks: 32 rows x 8 chunks of 8 bf16
      uint32_t off = (uint32_t)((((size_t)(b * NSEQ + n0 + r)) * CDIM + h * HD + q * 8) * 2);
      async_copy_b128(kp, off, &kc[r][q * 8]);
      async_copy_b128(kn, off, &kc[r][64 + q * 8]);
    }
    #pragma unroll
    for (int c0 = 0; c0 < 2; ++c0) {
      int c = t + c0 * 256;         // 512 chunks: 32 rows x 16 chunks of 4 f32
      int r = c >> 4, q = c & 15;
      uint32_t off = (uint32_t)(((size_t)(b * NSEQ + n0 + r) * CDIM + h * HD + q * 4) * 4);
      async_copy_b128(v, off, &vv[r][q * 4]);
    }
    wait_asynccnt<0>();
    __syncthreads();
    for (int n = 0; n < 32; ++n) {
      float kval = bf2f(kc[n][kk]);
      if (vb == 0) ks += kval;
      #pragma unroll
      for (int j = 0; j < 32; ++j) acc[j] = fmaf(kval, vv[n][vb + j], acc[j]);
    }
    __syncthreads();
  }

  #pragma unroll
  for (int j = 0; j < 32; ++j) kvs[kk][vb + j] = acc[j];
  if (vb == 0) ksum[kk] = ks;
  __syncthreads();

  const float invn = 1.f / (float)NSEQ;
  bf16* M = Mt + (size_t)bh * 80 * 128;
  for (int i = t; i < 80 * 128; i += 256) {
    int colv = i >> 7;      // 0..79
    int k    = i & 127;
    float val = 0.f;
    if      (colv < 32) val = kvs[k][colv] * invn;                    // kv1
    else if (colv < 64) val = kvs[(k + 64) & 127][colv] * invn;       // kv2 row-swapped
    else if (colv == 64) val = ksum[k] * invn;                        // k_mean
    else if (colv == 65) val = ksum[(k + 64) & 127] * invn;           // swapped mean
    M[i] = f2bf(val);
  }
}

// ---------------- batched attention GEMM: y = q_sim @ Mt^T, fused z ----------
__global__ __launch_bounds__(256) void attn_gemm_k(const bf16* __restrict__ qp,
                                                   const bf16* __restrict__ qn,
                                                   const bf16* __restrict__ Mt,
                                                   float* __restrict__ xa) {
  __shared__ __attribute__((aligned(16))) bf16 As[2][128][40];
  __shared__ __attribute__((aligned(16))) bf16 Bs[80][136];
  __shared__ float denS[128];
  __shared__ float denO[128];

  const int bh = blockIdx.y;
  const int b = bh / NUM_HEADS, h = bh % NUM_HEADS;
  const int mBase = blockIdx.x * 128;          // row block within NSEQ
  const int t    = threadIdx.x;
  const int lane = t & 31;
  const int wave = t >> 5;
  const int lc   = lane & 15;
  const int klo  = (lane & 16) ? 8 : 0;
  const int kb   = (lane & 16) ? 16 : 0;

  const bf16* M = Mt + (size_t)bh * 80 * 128;
  // whole Bt (80x128) streamed once (5 async b128 per thread)
  #pragma unroll
  for (int c0 = 0; c0 < 5; ++c0) {
    int c = t + c0 * 256;
    int r = c >> 4, q = c & 15;
    async_copy_b128(M, (uint32_t)(((size_t)r * 128 + q * 8) * 2), &Bs[r][q * 8]);
  }
  // A tile issue: q_sim = [q_pos | q_neg] selected by uniform k0
  auto issue_tile = [&](int ki, int buf) {
    const int k0 = ki * 32;
    const bf16* Qsrc = (k0 < 64) ? qp : qn;
    const int kadj = k0 & 63;
    #pragma unroll
    for (int c0 = 0; c0 < 2; ++c0) {
      int c = t + c0 * 256;
      int row = c >> 2, q = c & 3;
      uint32_t off = (uint32_t)((((size_t)(b * NSEQ + mBase + row)) * CDIM +
                                 h * HD + kadj + q * 8) * 2);
      async_copy_b128(Qsrc, off, &As[buf][row][q * 8]);
    }
  };

  v8f acc[5] = {};
  issue_tile(0, 0);
  for (int ki = 0; ki < 4; ++ki) {
    const int buf = ki & 1;
    if (ki + 1 < 4) {
      issue_tile(ki + 1, buf ^ 1);
      wait_asynccnt<2>();
    } else {
      wait_asynccnt<0>();
    }
    __syncthreads();
    const int k0 = ki * 32;
    v16bf a = load_afrag(&As[buf][wave * 16 + lc][0], klo);
    #pragma unroll
    for (int j = 0; j < 5; ++j) {
      v16bf bb = load_bfrag(&Bs[j * 16 + lc][k0], kb);
      acc[j] = __builtin_amdgcn_wmma_f32_16x16x32_bf16(false, a, false, bb,
                                                       (short)0, acc[j], false, false);
    }
    __syncthreads();
  }

  // broadcast z denominators (tile 4 cols 64,65) through LDS
  if (lc < 2) {
    #pragma unroll
    for (int i = 0; i < 8; ++i) {
      int r = wave * 16 + ((lane >> 4) * 8) + i;
      if (lc == 0) denS[r] = acc[4][i];
      else         denO[r] = acc[4][i];
    }
  }
  __syncthreads();

  #pragma unroll
  for (int j = 0; j < 4; ++j) {
    int colv = j * 16 + lc;                    // 0..63 -> [x_sim | x_opp]
    #pragma unroll
    for (int i = 0; i < 8; ++i) {
      int r = wave * 16 + ((lane >> 4) * 8) + i;
      float den = ((colv < 32) ? denS[r] : denO[r]) + EPSV;
      int n = mBase + r;
      xa[((size_t)(b * NSEQ + n)) * CDIM + h * HD + colv] = acc[j][i] / den;
    }
  }
}

// ---------------- depthwise 5x5 conv + gating, bf16 activation out ----------
__global__ __launch_bounds__(256) void conv_combine_k(const float* __restrict__ v,
                                                      const float* __restrict__ xa,
                                                      const float* __restrict__ g,
                                                      const float* __restrict__ dwc_w,
                                                      const float* __restrict__ dwc_b,
                                                      bf16* __restrict__ act) {
  size_t idx = (size_t)blockIdx.x * 256 + threadIdx.x;
  if (idx >= (size_t)MROWS * CDIM) return;
  int c = (int)(idx % CDIM);
  size_t row = idx / CDIM;           // b*NSEQ + n
  int n = (int)(row & (NSEQ - 1));
  int b = (int)(row >> 12);
  int cc = c & (HD - 1);
  int y = n >> 6, x = n & 63;

  float s = dwc_b[cc];
  #pragma unroll
  for (int ky = 0; ky < 5; ++ky) {
    int yy = y + ky - 2;
    if (yy < 0 || yy > 63) continue;
    #pragma unroll
    for (int kx = 0; kx < 5; ++kx) {
      int xx = x + kx - 2;
      if (xx < 0 || xx > 63) continue;
      s += v[((size_t)(b * NSEQ) + yy * 64 + xx) * CDIM + c] * dwc_w[cc * 25 + ky * 5 + kx];
    }
  }
  act[idx] = f2bf((xa[idx] + s) * g[idx]);
}

// ---------------------------------------------------------------------------
extern "C" void kernel_launch(void* const* d_in, const int* in_sizes, int n_in,
                              void* d_out, int out_size, void* d_ws, size_t ws_size,
                              hipStream_t stream) {
  (void)in_sizes; (void)n_in; (void)out_size; (void)ws_size;

  const float* x       = (const float*)d_in[0];
  const float* Wqg     = (const float*)d_in[1];
  const float* Wkv     = (const float*)d_in[2];
  const float* Wproj   = (const float*)d_in[3];
  const float* bproj   = (const float*)d_in[4];
  const float* dwc_w   = (const float*)d_in[5];
  const float* dwc_b   = (const float*)d_in[6];
  const float* power_p = (const float*)d_in[7];
  const float* scale_p = (const float*)d_in[8];
  const float* pos_enc = (const float*)d_in[9];
  float* out = (float*)d_out;

  // bump allocator over workspace
  char* ws = (char*)d_ws;
  size_t off = 0;
  auto alloc = [&](size_t bytes) -> void* {
    void* p = ws + off;
    off = (off + bytes + 255) & ~(size_t)255;
    return p;
  };
  const size_t NE = (size_t)MROWS * CDIM;        // 25.17M elements
  bf16*  xb     = (bf16*)alloc(NE * 2);
  bf16*  WqgT   = (bf16*)alloc((size_t)2 * CDIM * CDIM * 2);
  bf16*  WkvT   = (bf16*)alloc((size_t)2 * CDIM * CDIM * 2);
  bf16*  WprojT = (bf16*)alloc((size_t)CDIM * CDIM * 2);
  bf16*  qp     = (bf16*)alloc(NE * 2);
  bf16*  qn     = (bf16*)alloc(NE * 2);
  bf16*  kp     = (bf16*)alloc(NE * 2);
  bf16*  kn     = (bf16*)alloc(NE * 2);
  float* gbuf   = (float*)alloc(NE * 4);
  float* vbuf   = (float*)alloc(NE * 4);
  bf16*  Mt     = (bf16*)alloc((size_t)BDIM * NUM_HEADS * 80 * 128 * 2);
  float* xabuf  = (float*)alloc(NE * 4);
  bf16*  act    = (bf16*)alloc(NE * 2);

  // 1) conversions
  cvt_f32_bf16_k<<<(unsigned)((NE + 255) / 256), 256, 0, stream>>>(x, xb, NE);
  transpose_f32_bf16_k<<<(unsigned)(((size_t)CDIM * 2 * CDIM + 255) / 256), 256, 0, stream>>>(
      Wqg, WqgT, CDIM, 2 * CDIM);
  transpose_f32_bf16_k<<<(unsigned)(((size_t)CDIM * 2 * CDIM + 255) / 256), 256, 0, stream>>>(
      Wkv, WkvT, CDIM, 2 * CDIM);
  transpose_f32_bf16_k<<<(unsigned)(((size_t)CDIM * CDIM + 255) / 256), 256, 0, stream>>>(
      Wproj, WprojT, CDIM, CDIM);

  // 2) x @ Wqg -> q_pos/q_neg (bf16) + g (fp32)
  GemmParams pqg = {};
  pqg.A = xb; pqg.Bt = WqgT; pqg.M = MROWS; pqg.K = CDIM; pqg.Ncols = 2 * CDIM;
  pqg.outPos = qp; pqg.outNeg = qn; pqg.outF = gbuf;
  pqg.scale_p = scale_p; pqg.power_p = power_p;
  gemm_bf16_wmma<0><<<dim3(MROWS / 128, (2 * CDIM) / 64), 256, 0, stream>>>(pqg);

  // 3) x @ Wkv -> k_pos/k_neg (bf16, +pos_enc) + v (fp32)
  GemmParams pkv = {};
  pkv.A = xb; pkv.Bt = WkvT; pkv.M = MROWS; pkv.K = CDIM; pkv.Ncols = 2 * CDIM;
  pkv.outPos = kp; pkv.outNeg = kn; pkv.outF = vbuf;
  pkv.scale_p = scale_p; pkv.power_p = power_p; pkv.pos_enc = pos_enc;
  gemm_bf16_wmma<1><<<dim3(MROWS / 128, (2 * CDIM) / 64), 256, 0, stream>>>(pkv);

  // 4) per-(b,h) kv1/kv2/k_mean reduction -> Mt
  kv_reduce_k<<<BDIM * NUM_HEADS, 256, 0, stream>>>(kp, kn, vbuf, Mt);

  // 5) batched attention GEMM + z normalization -> xa (fp32)
  attn_gemm_k<<<dim3(NSEQ / 128, BDIM * NUM_HEADS), 256, 0, stream>>>(qp, qn, Mt, xabuf);

  // 6) depthwise conv + gating -> bf16 activations
  conv_combine_k<<<(unsigned)((NE + 255) / 256), 256, 0, stream>>>(
      vbuf, xabuf, gbuf, dwc_w, dwc_b, act);

  // 7) final projection: act @ Wproj + bproj -> out (fp32)
  GemmParams pfin = {};
  pfin.A = act; pfin.Bt = WprojT; pfin.M = MROWS; pfin.K = CDIM; pfin.Ncols = CDIM;
  pfin.outFinal = out; pfin.bias = bproj;
  gemm_bf16_wmma<2><<<dim3(MROWS / 128, CDIM / 64), 256, 0, stream>>>(pfin);
}
